// RNNSequenceClassifier_43319040148145
// MI455X (gfx1250) — matrix operand order repaired and checked by reading, not compiled, not run on hardware
//
#include <hip/hip_runtime.h>
#include <hip/hip_bf16.h>

typedef __bf16 bf16;
typedef bf16 v8bf  __attribute__((ext_vector_type(8)));
typedef bf16 v16bf __attribute__((ext_vector_type(16)));
typedef float v8f  __attribute__((ext_vector_type(8)));

#define B_   64
#define S_   128
#define T_   32
#define IN_  300
#define KG_  100
#define H_   256
#define C_   3
#define KPAD 512      // IN_ + 2*KG_ = 500 -> padded to 512 (mult of 32)
#define G4   1024     // 4*H_
#define WENT_N 112    // 100 -> 112 (7 n-tiles)
#define WENT_K 224    // 200 -> 224 (7 k-steps)

// ---------------- WMMA fragment helpers (CDNA5 ISA 7.12.2 layouts) ----------

__device__ __forceinline__ v16bf cat8(v8bf lo, v8bf hi) {
  return __builtin_shufflevector(lo, hi, 0,1,2,3,4,5,6,7,8,9,10,11,12,13,14,15);
}

// A-matrix 16x32 bf16, row-major source with leading dim `ld`.
// lanes 0-15: M=lane,  elems 0-7 -> K=kb+0..7,  elems 8-15 -> K=kb+16..23
// lanes16-31: M=lane-16, elems 0-7 -> K=kb+8..15, elems 8-15 -> K=kb+24..31
__device__ __forceinline__ v16bf load_a_frag(const bf16* base, int ld, int m0,
                                             int kb, int lane) {
  int m  = m0 + (lane & 15);
  int kl = kb + ((lane >> 4) << 3);
  const bf16* p = base + m * ld + kl;
  v8bf lo = *(const v8bf*)p;
  v8bf hi = *(const v8bf*)(p + 16);
  return cat8(lo, hi);
}

// B-matrix 32x16 bf16 where source holds B^T row-major: rows are output cols n.
// lanes 0-15: K=kb+0..7,kb+16..23; lanes 16-31: K=kb+8..15,kb+24..31.
__device__ __forceinline__ v16bf load_b_frag(const bf16* base, int ld, int n0,
                                             int kb, int lane) {
  int n  = n0 + (lane & 15);
  int k0 = kb + ((lane >> 4) << 3);
  const bf16* p = base + n * ld + k0;
  v8bf lo = *(const v8bf*)p;
  v8bf hi = *(const v8bf*)(p + 16);
  return cat8(lo, hi);
}

__device__ __forceinline__ float sigmoidf_(float x) {
  return 1.0f / (1.0f + expf(-x));
}

// ------------------------- prep: bf16 weight packs --------------------------

__global__ void prep_kernel(const float* __restrict__ w_ih,
                            const float* __restrict__ w_hh,
                            const float* __restrict__ W_ent,
                            const float* __restrict__ b_ih,
                            const float* __restrict__ b_hh,
                            bf16* __restrict__ w_ih_bf, bf16* __restrict__ w_hh_bf,
                            bf16* __restrict__ w_ent_bf, float* __restrict__ biasSum) {
  int stride = gridDim.x * blockDim.x;
  for (int idx = blockIdx.x * blockDim.x + threadIdx.x; idx < G4 * KPAD; idx += stride) {
    int n = idx / KPAD, k = idx % KPAD;
    w_ih_bf[idx] = (k < IN_ + 2 * KG_) ? (bf16)w_ih[n * (IN_ + 2 * KG_) + k] : (bf16)0.0f;
    if (idx < G4 * H_) w_hh_bf[idx] = (bf16)w_hh[idx];
    if (idx < WENT_N * WENT_K) {
      int nn = idx / WENT_K, kk = idx % WENT_K;
      w_ent_bf[idx] = (nn < KG_ && kk < 2 * KG_) ? (bf16)W_ent[nn * 2 * KG_ + kk]
                                                 : (bf16)0.0f;
    }
    if (idx < G4) biasSum[idx] = b_ih[idx] + b_hh[idx];
  }
}

// --------------- kernel A: gathers + KG attention (WMMA) per token ----------
// one block per (b,s) token; 7 waves; builds bf16 row T[token, 0..511]

__global__ __launch_bounds__(224)
void kg_attention_kernel(const int* __restrict__ inputs, const int* __restrict__ triples,
                         const int* __restrict__ id2,
                         const float* __restrict__ word_emb,
                         const float* __restrict__ ent_emb,
                         const float* __restrict__ rel_emb,
                         const bf16* __restrict__ w_ent_bf,
                         bf16* __restrict__ Tbf) {
  __shared__ bf16  sA[T_][WENT_K];   // head_tail bf16, K-padded (A matrix)
  __shared__ float sRel[T_][KG_];    // tanh(rel)
  __shared__ float sHt[T_][WENT_N];  // tanh(head_tail @ W_ent^T)
  __shared__ float sAlpha[T_];
  __shared__ int   sTrip[T_][3];
  __shared__ int   sValid;

  const int token = blockIdx.x;
  const int tid = threadIdx.x;
  if (tid == 0) sValid = 0;
  if (tid < T_ * 3) ((int*)sTrip)[tid] = triples[token * T_ * 3 + tid];
  __syncthreads();

  // gather head/tail/rel embeddings
  for (int idx = tid; idx < T_ * KG_; idx += blockDim.x) {
    int t = idx / KG_, d = idx % KG_;
    sA[t][d]       = (bf16)ent_emb[sTrip[t][0] * KG_ + d];
    sA[t][KG_ + d] = (bf16)ent_emb[sTrip[t][1] * KG_ + d];
    sRel[t][d]     = tanhf(rel_emb[sTrip[t][2] * KG_ + d]);
  }
  for (int idx = tid; idx < T_ * (WENT_K - 2 * KG_); idx += blockDim.x) {
    int t = idx / (WENT_K - 2 * KG_), k = idx % (WENT_K - 2 * KG_);
    sA[t][2 * KG_ + k] = (bf16)0.0f;
  }
  if (tid < T_) {
    if (id2[token * T_ + tid] != -1) atomicOr(&sValid, 1);
  }
  __syncthreads();

  // WMMA: ht[32,112] = A[32,224] @ Went^T ; wave w owns n-tile w
  const int wv = tid >> 5, lane = tid & 31;
  const int n0 = wv * 16;
  v8f acc0 = {}, acc1 = {};
  for (int kk = 0; kk < WENT_K / 32; ++kk) {
    v16bf bF = load_b_frag(w_ent_bf, WENT_K, n0, kk * 32, lane);
    v16bf a0 = load_a_frag(&sA[0][0], WENT_K, 0,  kk * 32, lane);
    v16bf a1 = load_a_frag(&sA[0][0], WENT_K, 16, kk * 32, lane);
    acc0 = __builtin_amdgcn_wmma_f32_16x16x32_bf16(false, a0, false, bF, (short)0, acc0, false, false);
    acc1 = __builtin_amdgcn_wmma_f32_16x16x32_bf16(false, a1, false, bF, (short)0, acc1, false, false);
  }
  {
    int n  = n0 + (lane & 15);
    int mh = (lane >> 4) << 3;
    #pragma unroll
    for (int vi = 0; vi < 8; ++vi) {
      sHt[mh + vi][n]      = tanhf(acc0[vi]);
      sHt[16 + mh + vi][n] = tanhf(acc1[vi]);
    }
  }
  __syncthreads();

  // e[t] = sum_d ht[t,d] * tanh(rel[t,d]); softmax over triples
  if (tid < T_) {
    float e = 0.f;
    for (int d = 0; d < KG_; ++d) e += sHt[tid][d] * sRel[tid][d];
    sAlpha[tid] = e;
  }
  __syncthreads();
  if (tid == 0) {
    float mx = sAlpha[0];
    for (int t = 1; t < T_; ++t) mx = fmaxf(mx, sAlpha[t]);
    float sum = 0.f;
    for (int t = 0; t < T_; ++t) { float ex = expf(sAlpha[t] - mx); sAlpha[t] = ex; sum += ex; }
    float inv = 1.f / sum;
    for (int t = 0; t < T_; ++t) sAlpha[t] *= inv;
  }
  __syncthreads();

  // emit bf16 LSTM-input row: [word_emb | graph_embed | zero pad]
  const int wordIdx = inputs[token];
  const bool valid = (sValid != 0);
  for (int j = tid; j < KPAD; j += blockDim.x) {
    float v;
    if (j < IN_) {
      v = word_emb[wordIdx * IN_ + j];
    } else if (j < IN_ + 2 * KG_) {
      float g = 0.f;
      int k = j - IN_;
      if (valid)
        for (int t = 0; t < T_; ++t) g += sAlpha[t] * (float)sA[t][k];
      v = g;
    } else {
      v = 0.f;
    }
    Tbf[(long long)token * KPAD + j] = (bf16)v;
  }
}

// ----- kernel B: Gx[8192,1024] = T @ w_ih^T + (b_ih + b_hh)  (WMMA GEMM) ----
// each wave computes one 16x64 strip (4 n-tiles); 8 waves/block

__global__ __launch_bounds__(256)
void gates_x_gemm(const bf16* __restrict__ Tbf, const bf16* __restrict__ w_ih_bf,
                  const float* __restrict__ biasSum, float* __restrict__ Gx) {
  const int lane = threadIdx.x & 31;
  const int wv   = threadIdx.x >> 5;
  const int job  = blockIdx.x * 8 + wv;   // 512 mtiles * 16 nstrips = 8192 jobs
  const int mtile  = job >> 4;
  const int nstrip = job & 15;

  v8f acc[4] = {};
  for (int kk = 0; kk < KPAD / 32; ++kk) {
    v16bf a = load_a_frag(Tbf, KPAD, mtile * 16, kk * 32, lane);
    #pragma unroll
    for (int f = 0; f < 4; ++f) {
      v16bf b = load_b_frag(w_ih_bf, KPAD, nstrip * 64 + f * 16, kk * 32, lane);
      acc[f] = __builtin_amdgcn_wmma_f32_16x16x32_bf16(false, a, false, b, (short)0, acc[f], false, false);
    }
  }
  const int mh = mtile * 16 + ((lane >> 4) << 3);
  #pragma unroll
  for (int f = 0; f < 4; ++f) {
    int n = nstrip * 64 + f * 16 + (lane & 15);
    float bias = biasSum[n];
    #pragma unroll
    for (int vi = 0; vi < 8; ++vi)
      Gx[(long long)(mh + vi) * G4 + n] = acc[f][vi] + bias;
  }
}

// ---------------- kernel C: LSTM recurrence (persistent workgroup) ----------
// 32 waves; wave w: jt = w&15 owns hidden cols [jt*16, jt*16+16) across all 4
// gate quarters, mhalf = w>>4 owns batch rows [mhalf*32, mhalf*32+32).
// Each lane therefore holds i/f/g/o for the SAME (m, j) in its 4 accumulators.

__global__ __launch_bounds__(1024)
void lstm_kernel(const float* __restrict__ Gx, const bf16* __restrict__ w_hh_bf,
                 const int* __restrict__ lengths, float* __restrict__ outseq) {
  __shared__ bf16 sH[B_][H_];   // 64x256 bf16 hidden state
  const int tid = threadIdx.x, lane = tid & 31, wv = tid >> 5;
  const int jt = wv & 15, mhalf = wv >> 4;

  for (int i = tid; i < B_ * H_; i += 1024) ((bf16*)sH)[i] = (bf16)0.0f;
  float creg[2][8];
  #pragma unroll
  for (int mt = 0; mt < 2; ++mt)
    #pragma unroll
    for (int vi = 0; vi < 8; ++vi) creg[mt][vi] = 0.f;
  __syncthreads();

  const int nj = jt * 16 + (lane & 15);              // hidden col j
  const int mbase = mhalf * 32 + ((lane >> 4) << 3); // batch row base

  for (int s = 0; s < S_; ++s) {
    v8f acc[2][4] = {};
    for (int kk = 0; kk < H_ / 32; ++kk) {
      v16bf a0 = load_a_frag(&sH[0][0], H_, mhalf * 32,      kk * 32, lane);
      v16bf a1 = load_a_frag(&sH[0][0], H_, mhalf * 32 + 16, kk * 32, lane);
      #pragma unroll
      for (int q = 0; q < 4; ++q) {
        v16bf b = load_b_frag(w_hh_bf, H_, q * H_ + jt * 16, kk * 32, lane);
        acc[0][q] = __builtin_amdgcn_wmma_f32_16x16x32_bf16(false, a0, false, b, (short)0, acc[0][q], false, false);
        acc[1][q] = __builtin_amdgcn_wmma_f32_16x16x32_bf16(false, a1, false, b, (short)0, acc[1][q], false, false);
      }
    }
    __syncthreads();   // all h reads done before overwrite
    #pragma unroll
    for (int mt = 0; mt < 2; ++mt) {
      #pragma unroll
      for (int vi = 0; vi < 8; ++vi) {
        int m = mbase + mt * 16 + vi;
        long long rowoff = (long long)(m * S_ + s) * G4;
        float ig = acc[mt][0][vi] + Gx[rowoff + 0 * H_ + nj];
        float fg = acc[mt][1][vi] + Gx[rowoff + 1 * H_ + nj];
        float gg = acc[mt][2][vi] + Gx[rowoff + 2 * H_ + nj];
        float og = acc[mt][3][vi] + Gx[rowoff + 3 * H_ + nj];
        float c = sigmoidf_(fg) * creg[mt][vi] + sigmoidf_(ig) * tanhf(gg);
        float h = sigmoidf_(og) * tanhf(c);
        creg[mt][vi] = c;
        sH[m][nj] = (bf16)h;
        outseq[(long long)(m * S_ + s) * H_ + nj] = (s < lengths[m]) ? h : 0.0f;
      }
    }
    __syncthreads();   // h writes visible before next step
  }
}

// -------------- kernel D: masked attention pooling + classifier -------------

__global__ __launch_bounds__(128)
void attn_out_kernel(const float* __restrict__ outseq, const float* __restrict__ W_att,
                     const float* __restrict__ b_att, const float* __restrict__ W_out,
                     const float* __restrict__ b_out, float* __restrict__ out) {
  __shared__ float sLogit[S_];
  __shared__ float sP[S_];
  __shared__ float sEnc[H_];
  const int b = blockIdx.x, tid = threadIdx.x;

  {
    const float* row = outseq + (long long)(b * S_ + tid) * H_;
    float l = b_att[0];
    for (int h = 0; h < H_; ++h) l += row[h] * W_att[h];
    sLogit[tid] = l;
  }
  __syncthreads();
  if (tid == 0) {
    float z[S_];
    float mx = -1e30f;
    for (int s = 0; s < S_; ++s) {
      float m = (sLogit[s] != 0.0f) ? 1.0f : 0.0f;
      z[s] = sLogit[s] * m;
      mx = fmaxf(mx, z[s]);
    }
    float sum = 0.f;
    for (int s = 0; s < S_; ++s) { z[s] = expf(z[s] - mx); sum += z[s]; }
    float psum = 0.f;
    for (int s = 0; s < S_; ++s) {
      float m = (sLogit[s] != 0.0f) ? 1.0f : 0.0f;
      z[s] = (z[s] / sum) * m;
      psum += z[s];
    }
    float inv = 1.0f / (psum + 1e-13f);
    for (int s = 0; s < S_; ++s) sP[s] = z[s] * inv;
  }
  __syncthreads();
  for (int h = tid; h < H_; h += blockDim.x) {
    float e = 0.f;
    for (int s = 0; s < S_; ++s) e += sP[s] * outseq[(long long)(b * S_ + s) * H_ + h];
    sEnc[h] = e;
  }
  __syncthreads();
  if (tid == 0) {
    float lg[C_];
    float mx = -1e30f;
    for (int c = 0; c < C_; ++c) {
      float v = b_out[c];
      for (int h = 0; h < H_; ++h) v += sEnc[h] * W_out[c * H_ + h];
      lg[c] = v;
      mx = fmaxf(mx, v);
    }
    float sum = 0.f;
    for (int c = 0; c < C_; ++c) sum += expf(lg[c] - mx);
    float lse = mx + logf(sum);
    for (int c = 0; c < C_; ++c) out[b * C_ + c] = lg[c] - lse;
  }
  out[C_ * B_ + b * S_ + tid] = sP[tid];
}

// ---------------------------------------------------------------------------

extern "C" void kernel_launch(void* const* d_in, const int* in_sizes, int n_in,
                              void* d_out, int out_size, void* d_ws, size_t ws_size,
                              hipStream_t stream) {
  (void)in_sizes; (void)n_in; (void)out_size; (void)ws_size;
  const int*   inputs   = (const int*)d_in[0];
  const int*   triples  = (const int*)d_in[1];
  const int*   lengths  = (const int*)d_in[2];
  const int*   id2      = (const int*)d_in[3];
  const float* word_emb = (const float*)d_in[4];
  const float* ent_emb  = (const float*)d_in[5];
  const float* rel_emb  = (const float*)d_in[6];
  const float* W_ent    = (const float*)d_in[7];
  const float* w_ih     = (const float*)d_in[8];
  const float* w_hh     = (const float*)d_in[9];
  const float* b_ih     = (const float*)d_in[10];
  const float* b_hh     = (const float*)d_in[11];
  const float* W_att    = (const float*)d_in[12];
  const float* b_att    = (const float*)d_in[13];
  const float* W_out    = (const float*)d_in[14];
  const float* b_out    = (const float*)d_in[15];
  float* out = (float*)d_out;

  char* ws = (char*)d_ws;
  bf16*  Tbf      = (bf16*)(ws);                 // 8192*512*2  = 8,388,608
  bf16*  w_ih_bf  = (bf16*)(ws + 8388608);       // 1024*512*2  = 1,048,576
  bf16*  w_hh_bf  = (bf16*)(ws + 9437184);       // 1024*256*2  =   524,288
  bf16*  w_ent_bf = (bf16*)(ws + 9961472);       // 112*224*2   =    50,176
  float* biasSum  = (float*)(ws + 10011648);     // 1024*4      =     4,096
  float* Gx       = (float*)(ws + 10015744);     // 8192*1024*4 = 33,554,432
  float* outseq   = (float*)(ws + 43570176);     // 8192*256*4  =  8,388,608

  hipLaunchKernelGGL(prep_kernel, dim3(512), dim3(256), 0, stream,
                     w_ih, w_hh, W_ent, b_ih, b_hh,
                     w_ih_bf, w_hh_bf, w_ent_bf, biasSum);
  hipLaunchKernelGGL(kg_attention_kernel, dim3(B_ * S_), dim3(224), 0, stream,
                     inputs, triples, id2, word_emb, ent_emb, rel_emb, w_ent_bf, Tbf);
  hipLaunchKernelGGL(gates_x_gemm, dim3(1024), dim3(256), 0, stream,
                     Tbf, w_ih_bf, biasSum, Gx);
  hipLaunchKernelGGL(lstm_kernel, dim3(1), dim3(1024), 0, stream,
                     Gx, w_hh_bf, lengths, outseq);
  hipLaunchKernelGGL(attn_out_kernel, dim3(B_), dim3(128), 0, stream,
                     outseq, W_att, b_att, W_out, b_out, out);
}
